// BareMultiHeadAttention_49237505081565
// MI455X (gfx1250) — matrix operand order repaired and checked by reading, not compiled, not run on hardware
//
#include <hip/hip_runtime.h>
#include <hip/hip_bf16.h>

// MI455X / gfx1250 multi-head attention (B=2, H=16, S=2048, W=64, D=1024)
// K0: f32 -> bf16 pre-convert of x and W (one pass, HBM-bound)
// K1: QKV projection; async-DMA double-buffered LDS tiles -> v_wmma bf16/f32
//     Q,K stored (b,h,s,w); V stored transposed (b,h,w,s) for fast PV frags
// K2: flash attention; async-DMA K/V^T chunks -> v_wmma bf16/f32

typedef __attribute__((ext_vector_type(16))) __bf16 bf16x16;
typedef __attribute__((ext_vector_type(8)))  __bf16 bf16x8;
typedef __attribute__((ext_vector_type(4)))  __bf16 bf16x4;
typedef __attribute__((ext_vector_type(8)))  float  f32x8;
typedef __attribute__((ext_vector_type(4)))  int    v4i;

typedef __attribute__((address_space(1))) v4i gv4i;   // global int4
typedef __attribute__((address_space(3))) v4i sv4i;   // LDS int4

#define BATCH   2
#define HEADS   16
#define SEQ     2048
#define WDIM    64
#define DMODEL  1024
#define TKEYS   32
#define KT      32
#define QSZ     ((size_t)BATCH * HEADS * SEQ * WDIM)     // 4,194,304 per matrix

// ws layout (bf16 elements)
#define XB_OFF   ((size_t)0)                              // 4096*1024
#define WB_OFF   (XB_OFF + (size_t)BATCH * SEQ * DMODEL)  // 3 x 1024*1024
#define QKV_OFF  (WB_OFF + (size_t)3 * DMODEL * DMODEL)   // 3 x QSZ

// 16-byte async DMA: global -> LDS (one lane = one b128), tracked by ASYNCcnt
__device__ __forceinline__ void async_copy16(const __bf16* g, __bf16* l) {
  __builtin_amdgcn_global_load_async_to_lds_b128(
      (gv4i*)(g), (sv4i*)(l), 0, 0);
}

// ---------------------------------------------------------------------------
// Kernel 0: f32 -> bf16 (4 elements/thread)
// ---------------------------------------------------------------------------
__global__ __launch_bounds__(256) void cvt_bf16_kernel(
    const float* __restrict__ src, __bf16* __restrict__ dst, int n4)
{
  const int i = blockIdx.x * 256 + threadIdx.x;
  if (i < n4) {
    const float4 v = ((const float4*)src)[i];
    bf16x4 o;
    o.x = (__bf16)v.x; o.y = (__bf16)v.y; o.z = (__bf16)v.z; o.w = (__bf16)v.w;
    ((bf16x4*)dst)[i] = o;
  }
}

// ---------------------------------------------------------------------------
// Kernel 1: Q/K/V = x @ W^T + b -> bf16
// grid: (4096/256, 16, 3)  block: 256 (8 waves)
// Block tile: 256 rows x 64 cols; each wave 32 rows x 64 cols (8 WMMAs/k-step)
// ---------------------------------------------------------------------------
__global__ __launch_bounds__(256) void qkv_gemm_kernel(
    const __bf16* __restrict__ xb, const __bf16* __restrict__ wb,
    const float* __restrict__ bq, const float* __restrict__ bk,
    const float* __restrict__ bv, __bf16* __restrict__ qkv)
{
  const int tid  = threadIdx.x;
  const int lane = tid & 31;
  const int wave = tid >> 5;
  const int mrow = lane & 15;
  const int hi   = lane >> 4;
  const int koff = hi * 8;
  const int bko  = hi * 16;

  const int rowBlk  = blockIdx.x * 256;
  const int h       = blockIdx.y;
  const int colBase = h * WDIM;
  const int mat     = blockIdx.z;

  const __bf16* W    = wb + (size_t)mat * DMODEL * DMODEL;
  const float*  bias = (mat == 0) ? bq : ((mat == 1) ? bk : bv);
  __bf16* out = qkv + (size_t)mat * QSZ;

  __shared__ __align__(16) __bf16 ldsA[2][256 * KT];   // 2 x 16 KB
  __shared__ __align__(16) __bf16 ldsB[2][WDIM * KT];  // 2 x 4 KB

  auto stage = [&](int ti, int buf) {
    const int k0 = ti * KT;
    // A tile: 256 rows x 32 -> 1024 b128 chunks, 4 per thread
#pragma unroll
    for (int c = 0; c < 4; ++c) {
      const int chunk = tid + c * 256;
      const int row = chunk >> 2, part = chunk & 3;
      async_copy16(xb + (size_t)(rowBlk + row) * DMODEL + k0 + part * 8,
                   &ldsA[buf][row * KT + part * 8]);
    }
    // B tile: 64 rows (output cols) x 32 -> 256 b128 chunks, 1 per thread
    {
      const int row = tid >> 2, part = tid & 3;
      async_copy16(W + (size_t)(colBase + row) * DMODEL + k0 + part * 8,
                   &ldsB[buf][row * KT + part * 8]);
    }
  };

  f32x8 acc[2][4];
#pragma unroll
  for (int s = 0; s < 2; ++s)
#pragma unroll
    for (int t = 0; t < 4; ++t)
#pragma unroll
      for (int r = 0; r < 8; ++r) acc[s][t][r] = 0.0f;

  stage(0, 0);
  __builtin_amdgcn_s_wait_asynccnt(0);
  __syncthreads();

  const int NT = DMODEL / KT;   // 32
  for (int ti = 0; ti < NT; ++ti) {
    const int cur = ti & 1;
    if (ti + 1 < NT) stage(ti + 1, cur ^ 1);   // DMA overlaps this tile's WMMAs

    bf16x16 a[2];
#pragma unroll
    for (int s = 0; s < 2; ++s) {
      const __bf16* ar = &ldsA[cur][(wave * 32 + s * 16 + mrow) * KT];
#pragma unroll
      for (int j = 0; j < 8; ++j) {
        a[s][j]     = ar[2 * koff + j];
        a[s][8 + j] = ar[2 * koff + 8 + j];
      }
    }
#pragma unroll
    for (int t = 0; t < 4; ++t) {
      const __bf16* br = &ldsB[cur][(t * 16 + mrow) * KT + bko];
      bf16x16 bfrag;
#pragma unroll
      for (int j = 0; j < 16; ++j) bfrag[j] = br[j];
#pragma unroll
      for (int s = 0; s < 2; ++s)
        acc[s][t] = __builtin_amdgcn_wmma_f32_16x16x32_bf16(
            false, a[s], false, bfrag, (short)0, acc[s][t], false, false);
    }

    __builtin_amdgcn_s_wait_asynccnt(0);
    __syncthreads();
  }

  // epilogue: + bias; Q,K -> (b,h,s,w); V -> (b,h,w,s) transposed
#pragma unroll
  for (int s = 0; s < 2; ++s) {
    const int row0 = rowBlk + wave * 32 + s * 16;
    if (mat != 2) {
#pragma unroll
      for (int t = 0; t < 4; ++t) {
        const int w = t * 16 + mrow;
        const float bv_ = bias[colBase + w];
#pragma unroll
        for (int r = 0; r < 8; ++r) {
          const int m    = r + hi * 8;
          const int srow = row0 + m;
          const int bb   = srow >> 11;
          const int ss   = srow & (SEQ - 1);
          out[((((size_t)bb * HEADS + h) * SEQ) + ss) * WDIM + w] =
              (__bf16)(acc[s][t][r] + bv_);
        }
      }
    } else {
      const int sbase = row0 + hi * 8;          // 8 consecutive s per lane
      const int bb = sbase >> 11;
      const int ss = sbase & (SEQ - 1);
#pragma unroll
      for (int t = 0; t < 4; ++t) {
        const int w = t * 16 + mrow;
        const float bv_ = bias[colBase + w];
        bf16x8 v;
#pragma unroll
        for (int r = 0; r < 8; ++r) v[r] = (__bf16)(acc[s][t][r] + bv_);
        *(bf16x8*)(out + ((((size_t)bb * HEADS + h) * WDIM) + w) * SEQ + ss) = v;
      }
    }
  }
}

// ---------------------------------------------------------------------------
// Kernel 2: flash attention per (b,h).  grid: (S/128, B*H)  block: 256
// ---------------------------------------------------------------------------
__global__ __launch_bounds__(256) void attn_kernel(
    const __bf16* __restrict__ qkv,
    const float*  __restrict__ mask,
    float* __restrict__ out)
{
  const int tid  = threadIdx.x;
  const int lane = tid & 31;
  const int wave = tid >> 5;
  const int mrow = lane & 15;
  const int hi   = lane >> 4;
  const int koff = hi * 8;
  const int bko  = hi * 16;

  const int bh = blockIdx.y;
  const int b  = bh >> 4;
  const int h  = bh & 15;
  const int qBase = blockIdx.x * 128 + wave * 16;

  const __bf16* Q  = qkv +           (size_t)bh * SEQ * WDIM;
  const __bf16* K  = qkv + QSZ     + (size_t)bh * SEQ * WDIM;
  const __bf16* VT = qkv + 2 * QSZ + (size_t)bh * WDIM * SEQ;  // (w,s) layout

  __shared__ __align__(16) __bf16 ldsK [2][TKEYS * WDIM];   // 2 x 4 KB
  __shared__ __align__(16) __bf16 ldsVT[2][WDIM * TKEYS];   // 2 x 4 KB
  __shared__ float  ldsS[8][16 * TKEYS];                    // 16 KB
  __shared__ __bf16 ldsP[8][16 * TKEYS];                    // 8 KB
  __shared__ float  ldsM[8][16], ldsL[8][16], ldsA8[8][16];

  auto stageKV = [&](int t0, int buf) {
    // K chunk: 32 keys x 64 -> contiguous rows
    async_copy16(K + (size_t)t0 * WDIM + tid * 8, &ldsK[buf][tid * 8]);
    // V^T chunk: 64 w-rows x 32 keys, each row contiguous in s
    {
      const int row = tid >> 2, part = tid & 3;
      async_copy16(VT + (size_t)row * SEQ + t0 + part * 8,
                   &ldsVT[buf][row * TKEYS + part * 8]);
    }
  };

  // Persistent Q A-frags (two 32-wide halves of the 64-dim head)
  bf16x16 qa[2];
  {
    const __bf16* qrow = Q + (size_t)(qBase + mrow) * WDIM;
#pragma unroll
    for (int half = 0; half < 2; ++half)
#pragma unroll
      for (int j = 0; j < 8; ++j) {
        qa[half][j]     = qrow[half * 32 + 2 * koff + j];
        qa[half][8 + j] = qrow[half * 32 + 2 * koff + 8 + j];
      }
  }

  f32x8 acc[4];
#pragma unroll
  for (int t = 0; t < 4; ++t)
#pragma unroll
    for (int r = 0; r < 8; ++r) acc[t][r] = 0.0f;

  if (lane < 16) { ldsM[wave][lane] = -1e30f; ldsL[wave][lane] = 0.0f; }

  stageKV(0, 0);
  __builtin_amdgcn_s_wait_asynccnt(0);
  __syncthreads();

  const int NC = SEQ / TKEYS;   // 64
  for (int ci = 0; ci < NC; ++ci) {
    const int cur = ci & 1;
    const int t0  = ci * TKEYS;
    if (ci + 1 < NC) stageKV(t0 + TKEYS, cur ^ 1);   // DMA overlaps compute

    // scores = Q . K^T  (2 key tiles of 16, contraction over w in 2 halves)
    f32x8 sc[2];
#pragma unroll
    for (int nt = 0; nt < 2; ++nt) {
      f32x8 c;
#pragma unroll
      for (int r = 0; r < 8; ++r) c[r] = 0.0f;
#pragma unroll
      for (int half = 0; half < 2; ++half) {
        const __bf16* kr = &ldsK[cur][(nt * 16 + mrow) * WDIM + half * 32 + bko];
        bf16x16 bfrag;
#pragma unroll
        for (int j = 0; j < 16; ++j) bfrag[j] = kr[j];
        c = __builtin_amdgcn_wmma_f32_16x16x32_bf16(
            false, qa[half], false, bfrag, (short)0, c, false, false);
      }
      sc[nt] = c;
    }

    // scale + mask, scatter to per-wave score tile (16 rows x 32 keys)
#pragma unroll
    for (int nt = 0; nt < 2; ++nt) {
      const int tloc = nt * 16 + mrow;
      const float mk = mask[(size_t)b * SEQ + t0 + tloc];
      const float madd = -10000.0f * (1.0f - mk);
#pragma unroll
      for (int r = 0; r < 8; ++r) {
        const int m = r + hi * 8;
        ldsS[wave][m * TKEYS + tloc] = sc[nt][r] * 0.125f + madd;
      }
    }
    __syncthreads();

    // Online softmax: one lane per query row
    if (lane < 16) {
      const float mOld = ldsM[wave][lane];
      const float* srow = &ldsS[wave][lane * TKEYS];
      float cm = mOld;
#pragma unroll
      for (int j = 0; j < TKEYS; ++j) cm = fmaxf(cm, srow[j]);
      const float alpha = __expf(mOld - cm);
      float lsum = ldsL[wave][lane] * alpha;
      __bf16* prow = &ldsP[wave][lane * TKEYS];
#pragma unroll
      for (int j = 0; j < TKEYS; ++j) {
        const float p = __expf(srow[j] - cm);
        lsum += p;
        prow[j] = (__bf16)p;
      }
      ldsM[wave][lane] = cm;
      ldsL[wave][lane] = lsum;
      ldsA8[wave][lane] = alpha;
    }
    __syncthreads();

    // Rescale running accumulator
    float al[8];
#pragma unroll
    for (int r = 0; r < 8; ++r) al[r] = ldsA8[wave][r + hi * 8];
#pragma unroll
    for (int nt = 0; nt < 4; ++nt)
#pragma unroll
      for (int r = 0; r < 8; ++r) acc[nt][r] *= al[r];

    // Regather P as A-frag
    bf16x16 pa;
    {
      const __bf16* prow = &ldsP[wave][mrow * TKEYS];
#pragma unroll
      for (int j = 0; j < 8; ++j) {
        pa[j]     = prow[2 * koff + j];
        pa[8 + j] = prow[2 * koff + 8 + j];
      }
    }

    // acc += P . V  (4 w-tiles of 16); V^T rows make B-frags contiguous
#pragma unroll
    for (int nt = 0; nt < 4; ++nt) {
      const __bf16* vr = &ldsVT[cur][(nt * 16 + mrow) * TKEYS + bko];
      bf16x16 bfrag;
#pragma unroll
      for (int j = 0; j < 16; ++j) bfrag[j] = vr[j];
      acc[nt] = __builtin_amdgcn_wmma_f32_16x16x32_bf16(
          false, pa, false, bfrag, (short)0, acc[nt], false, false);
    }

    __builtin_amdgcn_s_wait_asynccnt(0);
    __syncthreads();
  }

  // Finalize: /l, merged-head f32 store (B,S,H*W)
  float li[8];
#pragma unroll
  for (int r = 0; r < 8; ++r) li[r] = 1.0f / ldsL[wave][r + hi * 8];
#pragma unroll
  for (int nt = 0; nt < 4; ++nt) {
    const int w = nt * 16 + mrow;
#pragma unroll
    for (int r = 0; r < 8; ++r) {
      const int m = r + hi * 8;
      const int s = qBase + m;
      out[(((size_t)b * SEQ + s) * DMODEL) + h * WDIM + w] = acc[nt][r] * li[r];
    }
  }
}

// ---------------------------------------------------------------------------
extern "C" void kernel_launch(void* const* d_in, const int* in_sizes, int n_in,
                              void* d_out, int out_size, void* d_ws, size_t ws_size,
                              hipStream_t stream) {
  (void)in_sizes; (void)n_in; (void)out_size; (void)ws_size;
  const float* x  = (const float*)d_in[0];
  const float* mk = (const float*)d_in[1];
  const float* Wq = (const float*)d_in[2];
  const float* bq = (const float*)d_in[3];
  const float* Wk = (const float*)d_in[4];
  const float* bk = (const float*)d_in[5];
  const float* Wv = (const float*)d_in[6];
  const float* bv = (const float*)d_in[7];
  float* out = (float*)d_out;
  __bf16* ws = (__bf16*)d_ws;   // xb | Wb(q,k,v) | Q|K|V  (~38 MB bf16)

  __bf16* xb  = ws + XB_OFF;
  __bf16* wb  = ws + WB_OFF;
  __bf16* qkv = ws + QKV_OFF;

  // K0: one-pass f32->bf16
  {
    const int nx4 = BATCH * SEQ * DMODEL / 4;      // 1,048,576
    const int nw4 = DMODEL * DMODEL / 4;           //   262,144
    cvt_bf16_kernel<<<nx4 / 256, 256, 0, stream>>>(x, xb, nx4);
    cvt_bf16_kernel<<<nw4 / 256, 256, 0, stream>>>(Wq, wb + 0 * (size_t)DMODEL * DMODEL, nw4);
    cvt_bf16_kernel<<<nw4 / 256, 256, 0, stream>>>(Wk, wb + 1 * (size_t)DMODEL * DMODEL, nw4);
    cvt_bf16_kernel<<<nw4 / 256, 256, 0, stream>>>(Wv, wb + 2 * (size_t)DMODEL * DMODEL, nw4);
  }

  dim3 g1(BATCH * SEQ / 256, DMODEL / WDIM, 3);    // (16,16,3)
  qkv_gemm_kernel<<<g1, 256, 0, stream>>>(xb, wb, bq, bk, bv, qkv);

  dim3 g2(SEQ / 128, BATCH * HEADS, 1);            // (16,32)
  attn_kernel<<<g2, 256, 0, stream>>>(qkv, mk, out);
}
